// MAB_84026740179319
// MI455X (gfx1250) — compile-verified, matrix operand order checked by reading, not run on hardware
//
#include <hip/hip_runtime.h>

// ---------------------------------------------------------------- types ----
typedef __bf16 bf16;
typedef __attribute__((ext_vector_type(16))) __bf16 v16bf;
typedef __attribute__((ext_vector_type(8)))  __bf16 v8bf;
typedef __attribute__((ext_vector_type(8)))  float  v8f;

#define WMMA_BF16(a, b, c) \
  __builtin_amdgcn_wmma_f32_16x16x32_bf16(false, (a), false, (b), (short)0, (c), false, false)

// Problem constants (match reference)
constexpr int B_  = 4;
constexpr int N_  = 2048;
constexpr int M_  = 2048;
constexpr int D_  = 1024;   // DIM_Q = DIM_K = DIM_V
constexpr int H_  = 16;
constexpr int HD_ = 64;     // head dim

// ------------------------------------------------------------- fragments ---
// 16-bit A-fragment layout (ISA 05_wmma 7.12.2): lane = M row (mod 16),
// half (lane>>4) selects K+8; VGPR0..3 = K[8h .. 8h+7] (contiguous 16B),
// VGPR4..7 = K[16+8h .. 16+8h+7] (contiguous 16B).
// B-fragment of a 32x16 matrix stored as B^T row-major uses the identical map.
__device__ __forceinline__ v16bf load_tile16(const bf16* __restrict__ base, int ld) {
  const int lane = threadIdx.x & 31;
  const int row  = lane & 15;
  const int half = lane >> 4;
  const bf16* p = base + (size_t)row * ld + (half << 3);
  union { v16bf v; v8bf h[2]; } u;
  u.h[0] = *(const v8bf*)(p);
  u.h[1] = *(const v8bf*)(p + 16);
  return u.v;
}

// ------------------------------------------------------------ cvt kernel ---
__global__ void cvt_f32_to_bf16(const float* __restrict__ x, bf16* __restrict__ y, long n) {
  long i = (long)blockIdx.x * blockDim.x + threadIdx.x;
  const long stride = (long)gridDim.x * blockDim.x;
  for (; i < n; i += stride) y[i] = (bf16)x[i];
}

// ------------------------------------------------- projection GEMM kernel --
// Y = X @ W^T + bias ; 32 rows x 64 cols per wave (2 A-frags share each B-frag).
__global__ void proj_gemm_kernel(const bf16* __restrict__ X, const bf16* __restrict__ W,
                                 const float* __restrict__ bias,
                                 float* __restrict__ Yf, bf16* __restrict__ Ybf,
                                 bf16* __restrict__ Yt,
                                 int rows, int kdim, int ncols, int Mtr) {
  const int wave = threadIdx.x >> 5;
  const int lane = threadIdx.x & 31;
  const int row0 = blockIdx.x * 256 + wave * 32;
  const int col0 = blockIdx.y * 64;

  v8f acc[2][4] = {};
  for (int k = 0; k < kdim; k += 32) {
    const v16bf a0 = load_tile16(X + (size_t)row0 * kdim + k, kdim);
    const v16bf a1 = load_tile16(X + (size_t)(row0 + 16) * kdim + k, kdim);
#pragma unroll
    for (int t = 0; t < 4; ++t) {
      const v16bf bfr = load_tile16(W + (size_t)(col0 + 16 * t) * kdim + k, kdim);
      acc[0][t] = WMMA_BF16(a0, bfr, acc[0][t]);
      acc[1][t] = WMMA_BF16(a1, bfr, acc[1][t]);
    }
  }

  const int n16  = lane & 15;
  const int half = lane >> 4;
#pragma unroll
  for (int s = 0; s < 2; ++s) {
#pragma unroll
    for (int t = 0; t < 4; ++t) {
      const int col = col0 + 16 * t + n16;
      const float bv = bias[col];
#pragma unroll
      for (int r = 0; r < 8; ++r) {
        const int row = row0 + 16 * s + r + 8 * half;
        const float v = acc[s][t][r] + bv;
        const size_t idx = (size_t)row * ncols + col;
        if (Yf)  Yf[idx]  = v;
        if (Ybf) Ybf[idx] = (bf16)v;
        if (Yt)  Yt[((size_t)(row / Mtr) * ncols + col) * Mtr + (row % Mtr)] = (bf16)v;
      }
    }
  }
}

// ----------------------------------------------- async K/V tile staging ----
// K tile: 32 keys x 64 dims (32 rows x 128B). V tile (pre-transposed):
// 64 dims x 32 keys (64 rows x 64B). 256 threads, one b128 chunk each.
__device__ __forceinline__ void async_fill_kv(const bf16* __restrict__ Kg,
                                              const bf16* __restrict__ Vg,
                                              bf16* kbuf, bf16* vbuf) {
  const int t = threadIdx.x;
  {
    const int row = t >> 3, chunk = t & 7;                     // 32 x 8
    const unsigned long long ga =
        (unsigned long long)(uintptr_t)(Kg + (size_t)row * D_ + chunk * 8);
    const unsigned la = (unsigned)(uintptr_t)(kbuf + row * 64 + chunk * 8);
    asm volatile("global_load_async_to_lds_b128 %0, %1, off"
                 :: "v"(la), "v"(ga) : "memory");
  }
  {
    const int row = t >> 2, chunk = t & 3;                     // 64 x 4
    const unsigned long long ga =
        (unsigned long long)(uintptr_t)(Vg + (size_t)row * M_ + chunk * 8);
    const unsigned la = (unsigned)(uintptr_t)(vbuf + row * 32 + chunk * 8);
    asm volatile("global_load_async_to_lds_b128 %0, %1, off"
                 :: "v"(la), "v"(ga) : "memory");
  }
}

// --------------------------------------------------- flash attention core --
// Block = 8 waves, one (b,h) pair, 128 query rows. K/V tiles double-buffered
// in LDS via async loads and shared by all waves; P restaged via LDS.
__global__ void flash_attn_kernel(const bf16* __restrict__ Qpb, const bf16* __restrict__ Kpb,
                                  const bf16* __restrict__ Vt,  const float* __restrict__ Qpf,
                                  float* __restrict__ Of, bf16* __restrict__ Ob) {
  __shared__ bf16 kbufs[2][32 * 64];   // 2 x 4KB
  __shared__ bf16 vbufs[2][64 * 32];   // 2 x 4KB
  __shared__ bf16 ldsP[8][16 * 32];    // 8KB P staging (C-layout -> A-layout)

  const int wave = threadIdx.x >> 5;
  const int lane = threadIdx.x & 31;
  const int n16  = lane & 15;
  const int half = lane >> 4;

  const int b = blockIdx.x / H_;
  const int h = blockIdx.x % H_;
  const int q0 = blockIdx.y * 128 + wave * 16;

  const bf16* Kg = Kpb + (size_t)b * M_ * D_ + h * HD_;           // + j*D_ per step
  const bf16* Vg = Vt  + ((size_t)b * D_ + h * HD_) * M_;         // + j     per step

  const size_t qbase = ((size_t)b * N_ + q0) * D_ + h * HD_;
  const v16bf aq0 = load_tile16(Qpb + qbase,      D_);
  const v16bf aq1 = load_tile16(Qpb + qbase + 32, D_);

  v8f o0 = {}, o1 = {}, o2 = {}, o3 = {};
  float mrow[8], lrow[8];
#pragma unroll
  for (int r = 0; r < 8; ++r) { mrow[r] = -1e30f; lrow[r] = 0.0f; }

  const float scale = 0.03125f;          // 1/sqrt(DIM_V) = 1/32
  bf16* myP = &ldsP[wave][0];

  async_fill_kv(Kg, Vg, kbufs[0], vbufs[0]);   // prologue fill, buffer 0

  for (int j = 0; j < M_; j += 32) {
    const int cur = (j >> 5) & 1;
    if (j + 32 < M_) {
      async_fill_kv(Kg + (size_t)(j + 32) * D_, Vg + (j + 32),
                    kbufs[cur ^ 1], vbufs[cur ^ 1]);
      asm volatile("s_wait_asynccnt 2" ::: "memory");   // prior fill complete
    } else {
      asm volatile("s_wait_asynccnt 0" ::: "memory");
    }
    __syncthreads();                                    // all waves' fills visible

    const bf16* kb = kbufs[cur];
    const bf16* vb = vbufs[cur];

    // scores: 16 queries x 32 keys as two 16x16 C fragments (K=64 -> 2 WMMAs each)
    v8f s0 = {}, s1 = {};
    s0 = WMMA_BF16(aq0, load_tile16(kb,            64), s0);
    s0 = WMMA_BF16(aq1, load_tile16(kb + 32,       64), s0);
    s1 = WMMA_BF16(aq0, load_tile16(kb + 16 * 64,      64), s1);
    s1 = WMMA_BF16(aq1, load_tile16(kb + 16 * 64 + 32, 64), s1);

    // online softmax: rows of the C fragment live across the 16-lane halves
#pragma unroll
    for (int r = 0; r < 8; ++r) {
      const float a0 = s0[r] * scale;
      const float a1 = s1[r] * scale;
      float v = fmaxf(a0, a1);
#pragma unroll
      for (int off = 8; off >= 1; off >>= 1)
        v = fmaxf(v, __shfl_xor(v, off, 16));
      const float mnew = fmaxf(mrow[r], v);
      const float p0 = __expf(a0 - mnew);
      const float p1 = __expf(a1 - mnew);
      float ps = p0 + p1;
#pragma unroll
      for (int off = 8; off >= 1; off >>= 1)
        ps += __shfl_xor(ps, off, 16);
      const float ef = __expf(mrow[r] - mnew);
      lrow[r] = lrow[r] * ef + ps;
      mrow[r] = mnew;
      const int q = r + 8 * half;
      myP[q * 32 + n16]      = (bf16)p0;
      myP[q * 32 + 16 + n16] = (bf16)p1;
      o0[r] *= ef; o1[r] *= ef; o2[r] *= ef; o3[r] *= ef;
    }
    asm volatile("s_wait_dscnt 0" ::: "memory");        // wave-local P visibility

    const v16bf ap = load_tile16(myP, 32);              // P as 16x32 A-fragment
    o0 = WMMA_BF16(ap, load_tile16(vb,           32), o0);
    o1 = WMMA_BF16(ap, load_tile16(vb + 16 * 32, 32), o1);
    o2 = WMMA_BF16(ap, load_tile16(vb + 32 * 32, 32), o2);
    o3 = WMMA_BF16(ap, load_tile16(vb + 48 * 32, 32), o3);

    __syncthreads();   // all waves done reading buffers before they are refilled
  }

  // epilogue: normalize, add projected-Q residual, emit f32 + bf16 copies
#pragma unroll
  for (int r = 0; r < 8; ++r) {
    const float linv = 1.0f / lrow[r];
    const int q = q0 + r + 8 * half;
    const size_t rb = ((size_t)b * N_ + q) * D_ + h * HD_;
    const float v0 = o0[r] * linv + Qpf[rb + n16];
    const float v1 = o1[r] * linv + Qpf[rb + 16 + n16];
    const float v2 = o2[r] * linv + Qpf[rb + 32 + n16];
    const float v3 = o3[r] * linv + Qpf[rb + 48 + n16];
    Of[rb + n16]      = v0;  Ob[rb + n16]      = (bf16)v0;
    Of[rb + 16 + n16] = v1;  Ob[rb + 16 + n16] = (bf16)v1;
    Of[rb + 32 + n16] = v2;  Ob[rb + 32 + n16] = (bf16)v2;
    Of[rb + 48 + n16] = v3;  Ob[rb + 48 + n16] = (bf16)v3;
  }
}

// --------------------------------------------------- output GEMM + ReLU ----
// out = Xf + relu(Xb @ Wo^T + bo) ; 32 rows x 64 cols per wave.
__global__ void out_gemm_kernel(const bf16* __restrict__ Xb, const bf16* __restrict__ Wo,
                                const float* __restrict__ bo, const float* __restrict__ Xf,
                                float* __restrict__ out, int rows, int kdim) {
  const int wave = threadIdx.x >> 5;
  const int lane = threadIdx.x & 31;
  const int row0 = blockIdx.x * 256 + wave * 32;
  const int col0 = blockIdx.y * 64;

  v8f acc[2][4] = {};
  for (int k = 0; k < kdim; k += 32) {
    const v16bf a0 = load_tile16(Xb + (size_t)row0 * kdim + k, kdim);
    const v16bf a1 = load_tile16(Xb + (size_t)(row0 + 16) * kdim + k, kdim);
#pragma unroll
    for (int t = 0; t < 4; ++t) {
      const v16bf bfr = load_tile16(Wo + (size_t)(col0 + 16 * t) * kdim + k, kdim);
      acc[0][t] = WMMA_BF16(a0, bfr, acc[0][t]);
      acc[1][t] = WMMA_BF16(a1, bfr, acc[1][t]);
    }
  }

  const int n16  = lane & 15;
  const int half = lane >> 4;
#pragma unroll
  for (int s = 0; s < 2; ++s) {
#pragma unroll
    for (int t = 0; t < 4; ++t) {
      const int col = col0 + 16 * t + n16;
      const float bv = bo[col];
#pragma unroll
      for (int r = 0; r < 8; ++r) {
        const int row = row0 + 16 * s + r + 8 * half;
        const size_t idx = (size_t)row * kdim + col;
        out[idx] = Xf[idx] + fmaxf(acc[s][t][r] + bv, 0.0f);
      }
    }
  }
}

// ------------------------------------------------------------------ host ---
extern "C" void kernel_launch(void* const* d_in, const int* in_sizes, int n_in,
                              void* d_out, int out_size, void* d_ws, size_t ws_size,
                              hipStream_t stream) {
  const float* Q  = (const float*)d_in[0];
  const float* K  = (const float*)d_in[1];
  const float* Wq = (const float*)d_in[2];
  const float* bq = (const float*)d_in[3];
  const float* Wk = (const float*)d_in[4];
  const float* bk = (const float*)d_in[5];
  const float* Wv = (const float*)d_in[6];
  const float* bv = (const float*)d_in[7];
  const float* Wo = (const float*)d_in[8];
  const float* bo = (const float*)d_in[9];
  float* out = (float*)d_out;

  char* ws = (char*)d_ws;
  size_t off = 0;
  auto alloc = [&](size_t bytes) -> void* {
    void* p = ws + off;
    off += (bytes + 255) & ~(size_t)255;
    return p;
  };

  const size_t nQ = (size_t)B_ * N_ * D_;
  const size_t nK = (size_t)B_ * M_ * D_;
  const size_t nW = (size_t)D_ * D_;

  bf16*  Qb  = (bf16*)alloc(nQ * 2);
  bf16*  Kb  = (bf16*)alloc(nK * 2);
  bf16*  Wqb = (bf16*)alloc(nW * 2);
  bf16*  Wkb = (bf16*)alloc(nW * 2);
  bf16*  Wvb = (bf16*)alloc(nW * 2);
  bf16*  Wob = (bf16*)alloc(nW * 2);
  float* Qpf = (float*)alloc(nQ * 4);       // projected Q, f32 (residual)
  bf16*  Qpb = (bf16*)alloc(nQ * 2);        // projected Q, bf16
  bf16*  Kpb = (bf16*)alloc(nK * 2);        // projected K, bf16
  bf16*  Vtb = (bf16*)alloc(nK * 2);        // projected V, bf16, transposed [b][col][m]
  float* Of  = (float*)alloc(nQ * 4);       // attention out + residual, f32
  bf16*  Ob  = (bf16*)alloc(nQ * 2);
  (void)ws_size; (void)in_sizes; (void)n_in; (void)out_size;

  // 1) f32 -> bf16 casts
  cvt_f32_to_bf16<<<2048, 256, 0, stream>>>(Q,  Qb,  (long)nQ);
  cvt_f32_to_bf16<<<2048, 256, 0, stream>>>(K,  Kb,  (long)nK);
  cvt_f32_to_bf16<<<512,  256, 0, stream>>>(Wq, Wqb, (long)nW);
  cvt_f32_to_bf16<<<512,  256, 0, stream>>>(Wk, Wkb, (long)nW);
  cvt_f32_to_bf16<<<512,  256, 0, stream>>>(Wv, Wvb, (long)nW);
  cvt_f32_to_bf16<<<512,  256, 0, stream>>>(Wo, Wob, (long)nW);

  // 2) projections (WMMA bf16, f32 accumulate)
  dim3 gGemm(B_ * N_ / 256, D_ / 64);
  proj_gemm_kernel<<<gGemm, 256, 0, stream>>>(Qb, Wqb, bq, Qpf, Qpb, nullptr,
                                              B_ * N_, D_, D_, M_);
  proj_gemm_kernel<<<gGemm, 256, 0, stream>>>(Kb, Wkb, bk, nullptr, Kpb, nullptr,
                                              B_ * M_, D_, D_, M_);
  proj_gemm_kernel<<<gGemm, 256, 0, stream>>>(Kb, Wvb, bv, nullptr, nullptr, Vtb,
                                              B_ * M_, D_, D_, M_);

  // 3) fused flash attention per head + projected-Q residual
  dim3 gAttn(B_ * H_, N_ / 128);
  flash_attn_kernel<<<gAttn, 256, 0, stream>>>(Qpb, Kpb, Vtb, Qpf, Of, Ob);

  // 4) output projection + ReLU residual
  out_gemm_kernel<<<gGemm, 256, 0, stream>>>(Ob, Wob, bo, Of, out, B_ * N_, D_);
}
